// HybridTreeLSTM_26491358282207
// MI455X (gfx1250) — matrix operand order repaired and checked by reading, not compiled, not run on hardware
//
#include <hip/hip_runtime.h>
#include <hip/hip_bf16.h>

// ---------------- Problem constants ----------------
#define BATCH 16
#define SEQ   4096
#define HID   128
#define EDIM  100
#define KPAD  128          // E padded to 128 for WMMA K
#define NX    512          // 4 gates * H  (xg columns)
#define NU    640          // 5 gates * H  (pre columns)
#define MROWS (BATCH * SEQ)   // 65536

typedef __attribute__((ext_vector_type(16))) __bf16 v16bf;
typedef __attribute__((ext_vector_type(8)))  __bf16 v8bf;
typedef __attribute__((ext_vector_type(8)))  float  v8f;

__device__ __forceinline__ v8f zero8() {
    v8f z;
#pragma unroll
    for (int i = 0; i < 8; ++i) z[i] = 0.0f;
    return z;
}

__device__ __forceinline__ float sig_f(float x) {
    return 1.0f / (1.0f + __expf(-x));
}

// WMMA bf16 -> f32 accumulate, D = A*B + C
__device__ __forceinline__ v8f wmma_bf16(v16bf a, v16bf b, v8f c) {
    return __builtin_amdgcn_wmma_f32_16x16x32_bf16(
        /*neg_a=*/false, a, /*neg_b=*/false, b,
        /*c_mod=*/(short)0, c, /*reuse_a=*/false, /*reuse_b=*/false);
}

// A-fragment (16x32 bf16, MxK), row-major source, per-lane row pointer.
// lanes 0-15 hold K = k0+[0..15];
// lanes 16-31 hold K = k0+[16..31] (matches B fragment order).
__device__ __forceinline__ v16bf load_a_frag(const __bf16* __restrict__ row,
                                             int k0, int hsel) {
    const v8bf c0 = *(const v8bf*)(row + k0 + hsel * 16);
    const v8bf c1 = *(const v8bf*)(row + k0 + hsel * 16 + 8);
    v16bf r;
#pragma unroll
    for (int i = 0; i < 8; ++i) { r[i] = c0[i]; r[i + 8] = c1[i]; }
    return r;
}

// B-fragment (32x16 bf16, KxN) from weights stored [N][KPAD] row-major:
// lane (l&15) owns column n0+(l&15); lanes 0-15 hold K=k0..k0+15,
// lanes 16-31 hold K=k0+16..k0+31 -> one contiguous 32-byte load.
__device__ __forceinline__ v16bf load_b_frag(const __bf16* __restrict__ w,
                                             int n0, int k0, int lane) {
    const int n  = n0 + (lane & 15);
    const int kk = k0 + ((lane >> 4) & 1) * 16;
    return *(const v16bf*)(w + (size_t)n * KPAD + kk);
}

// ---------------- Kernel 0: fp32 weights -> bf16 [N][KPAD] ----------------
__global__ __launch_bounds__(256)
void prep_weights_kernel(const float* __restrict__ Wx_f,   // (4,H,E)
                         const float* __restrict__ Ul_f,   // (5,H,H)
                         const float* __restrict__ Ur_f,   // (5,H,H)
                         __bf16* __restrict__ Wx,          // [512][128]
                         __bf16* __restrict__ Ul,          // [640][128]
                         __bf16* __restrict__ Ur) {        // [640][128]
    const int t = blockIdx.x * 256 + threadIdx.x;
    const int WX_N = NX * KPAD;          // 65536
    const int UU_N = NU * KPAD;          // 81920
    if (t < WX_N) {
        const int n = t >> 7, k = t & 127;     // n = g*128 + h
        Wx[t] = (k < EDIM) ? (__bf16)Wx_f[(size_t)n * EDIM + k] : (__bf16)0.0f;
    } else if (t < WX_N + UU_N) {
        const int u = t - WX_N;                // (5,H,H) is already [n][k]
        Ul[u] = (__bf16)Ul_f[u];
    } else if (t < WX_N + 2 * UU_N) {
        const int u = t - WX_N - UU_N;
        Ur[u] = (__bf16)Ur_f[u];
    }
}

// ---------------- Kernel 1: embedding gather -> bf16 A [MROWS][KPAD] -------
__global__ __launch_bounds__(256)
void gather_embed_kernel(const float* __restrict__ emb,   // (V,E)
                         const int*   __restrict__ seq,   // (B,S) flat
                         __bf16* __restrict__ A) {
    const size_t t = (size_t)blockIdx.x * 256 + threadIdx.x;  // < MROWS*KPAD
    const int m = (int)(t >> 7);
    const int e = (int)(t & 127);
    const int tok = seq[m];
    A[t] = (e < EDIM) ? (__bf16)emb[(size_t)tok * EDIM + e] : (__bf16)0.0f;
}

// ---------------- Kernel 2: xg = A * Wx^T + b_x  (M=65536,N=512,K=128) -----
// block = 4 waves; wave owns a 16x64 tile; grid = (M/16, 2)
__global__ __launch_bounds__(128)
void xg_gemm_kernel(const __bf16* __restrict__ A,
                    const __bf16* __restrict__ Wx,
                    const float*  __restrict__ bx,        // (4,H) flat 512
                    float* __restrict__ xg) {             // [M][512]
    const int tid = threadIdx.x, lane = tid & 31, w = tid >> 5;
    const int m0     = blockIdx.x * 16;
    const int n_base = blockIdx.y * 256 + w * 64;
    const int hsel   = (lane >> 4) & 1;
    const __bf16* row = A + (size_t)(m0 + (lane & 15)) * KPAD;

    v8f acc[4];
#pragma unroll
    for (int j = 0; j < 4; ++j) acc[j] = zero8();

#pragma unroll
    for (int kk = 0; kk < 4; ++kk) {
        const int k0 = kk * 32;
        const v16bf a = load_a_frag(row, k0, hsel);
#pragma unroll
        for (int j = 0; j < 4; ++j)
            acc[j] = wmma_bf16(a, load_b_frag(Wx, n_base + j * 16, k0, lane), acc[j]);
    }

    // C/D layout: VGPR r -> row m0 + r + hsel*8, col n_base + j*16 + (lane&15)
#pragma unroll
    for (int j = 0; j < 4; ++j) {
        const int col = n_base + j * 16 + (lane & 15);
        const float bias = bx[col];
#pragma unroll
        for (int r = 0; r < 8; ++r) {
            const int rowi = m0 + r + hsel * 8;
            xg[(size_t)rowi * NX + col] = acc[j][r] + bias;
        }
    }
}

// ---------------- Kernel 3: level 0 (h=c=0 -> elementwise) -----------------
__global__ __launch_bounds__(256)
void level0_kernel(const float* __restrict__ xg,
                   const float* __restrict__ bl,          // (5,H) flat 640
                   const float* __restrict__ br,
                   __bf16* __restrict__ h0,               // [MROWS][128] bf16
                   float*  __restrict__ c0) {             // [MROWS][128] f32
    const size_t t = (size_t)blockIdx.x * 256 + threadIdx.x;  // < MROWS*128
    const int m = (int)(t >> 7);
    const int h = (int)(t & 127);
    const int xmap[5] = {0, 1, 1, 2, 3};
    float pre[5];
#pragma unroll
    for (int g = 0; g < 5; ++g)
        pre[g] = xg[(size_t)m * NX + xmap[g] * HID + h] + bl[g * HID + h] + br[g * HID + h];
    const float ig = sig_f(pre[0]);
    const float og = sig_f(pre[3]);
    const float u  = tanhf(pre[4]);
    const float c  = ig * u;            // f_l, f_r multiply zero cells
    const float hh = og * tanhf(c);
    h0[t] = (__bf16)hh;
    c0[t] = c;
}

// ---------------- Kernel 4: one tree level ---------------------------------
// block = 320 threads (10 waves); block owns 16 nodes x all 640 columns.
// pre = h_l*Ul^T + h_r*Ur^T  (WMMA) -> LDS -> gated epilogue with xg gather.
__global__ __launch_bounds__(320)
void tree_level_kernel(const __bf16* __restrict__ h_prev,   // [B*n_prev][128]
                       const float*  __restrict__ c_prev,
                       __bf16* __restrict__ h_out,          // [B*n][128]
                       float*  __restrict__ c_out,
                       const float*  __restrict__ xg,
                       const int*    __restrict__ dep,      // dep_tags row, len>=n
                       const __bf16* __restrict__ Ul,
                       const __bf16* __restrict__ Ur,
                       const float*  __restrict__ bl,
                       const float*  __restrict__ br,
                       int n, int n_prev,
                       float* __restrict__ final_out) {
    __shared__ float pre_lds[16 * NU];                      // 40 KB

    const int tid  = threadIdx.x;
    const int lane = tid & 31;
    const int w    = tid >> 5;               // 0..9 -> 64-col slice
    const int m0   = blockIdx.x * 16;        // global node tile (B*n is /16)
    const int hsel = (lane >> 4) & 1;

    // per-lane A rows: node i -> children rows 2j, 2j+1 of previous level
    const int i_row = m0 + (lane & 15);
    const int b_row = i_row / n;
    const int j_row = i_row - b_row * n;
    const __bf16* rowL = h_prev + (size_t)(b_row * n_prev + 2 * j_row) * HID;
    const __bf16* rowR = rowL + HID;

    v8f acc[4];
#pragma unroll
    for (int j = 0; j < 4; ++j) acc[j] = zero8();

#pragma unroll
    for (int kk = 0; kk < 4; ++kk) {
        const int k0 = kk * 32;
        const v16bf aL = load_a_frag(rowL, k0, hsel);
        const v16bf aR = load_a_frag(rowR, k0, hsel);
#pragma unroll
        for (int j = 0; j < 4; ++j) {
            const int n0 = w * 64 + j * 16;
            acc[j] = wmma_bf16(aL, load_b_frag(Ul, n0, k0, lane), acc[j]);
            acc[j] = wmma_bf16(aR, load_b_frag(Ur, n0, k0, lane), acc[j]);
        }
    }

    // spill pre to LDS in [node][gate*H] layout
#pragma unroll
    for (int j = 0; j < 4; ++j) {
        const int col = w * 64 + j * 16 + (lane & 15);
#pragma unroll
        for (int r = 0; r < 8; ++r)
            pre_lds[(r + hsel * 8) * NU + col] = acc[j][r];
    }
    __syncthreads();

    const int xmap[5] = {0, 1, 1, 2, 3};
    for (int e = tid; e < 16 * HID; e += 320) {
        const int node_r = e >> 7;
        const int h      = e & 127;
        const int i = m0 + node_r;
        const int b = i / n;
        const int j = i - b * n;
        const int d = dep[j];
        float pre[5];
#pragma unroll
        for (int g = 0; g < 5; ++g) {
            const size_t xoff = (((size_t)(b * SEQ + d)) * 4 + xmap[g]) * HID + h;
            pre[g] = pre_lds[node_r * NU + g * HID + h] + xg[xoff]
                   + bl[g * HID + h] + br[g * HID + h];
        }
        const float ig = sig_f(pre[0]);
        const float fl = sig_f(pre[1]);
        const float fr = sig_f(pre[2]);
        const float og = sig_f(pre[3]);
        const float u  = tanhf(pre[4]);
        const float cl = c_prev[(size_t)(b * n_prev + 2 * j) * HID + h];
        const float cr = c_prev[(size_t)(b * n_prev + 2 * j + 1) * HID + h];
        const float c  = ig * u + fl * cl + fr * cr;
        const float hh = og * tanhf(c);
        const size_t o = (size_t)(b * n + j) * HID + h;
        h_out[o] = (__bf16)hh;
        c_out[o] = c;
        if (final_out) {                       // last level: n==1, j==0
            final_out[b * HID + h]               = hh;
            final_out[BATCH * HID + b * HID + h] = c;
        }
    }
}

// ---------------- Host-side launcher ---------------------------------------
extern "C" void kernel_launch(void* const* d_in, const int* in_sizes, int n_in,
                              void* d_out, int out_size, void* d_ws, size_t ws_size,
                              hipStream_t stream) {
    (void)in_sizes; (void)n_in; (void)out_size; (void)ws_size;
    const float* emb    = (const float*)d_in[0];
    const float* Wx_f   = (const float*)d_in[1];
    const float* bx     = (const float*)d_in[2];
    const float* Ul_f   = (const float*)d_in[3];
    const float* bl     = (const float*)d_in[4];
    const float* Ur_f   = (const float*)d_in[5];
    const float* br     = (const float*)d_in[6];
    const int*   seq    = (const int*)d_in[7];
    const int*   deps   = (const int*)d_in[8];   // (12, 2048)
    float* out = (float*)d_out;

    char*  ws  = (char*)d_ws;
    size_t off = 0;
    auto alloc = [&](size_t bytes) -> void* {
        void* p = (void*)(ws + off);
        off += (bytes + 255) & ~(size_t)255;
        return p;
    };
    __bf16* Wx   = (__bf16*)alloc((size_t)NX * KPAD * 2);
    __bf16* Ul   = (__bf16*)alloc((size_t)NU * KPAD * 2);
    __bf16* Ur   = (__bf16*)alloc((size_t)NU * KPAD * 2);
    __bf16* Abf  = (__bf16*)alloc((size_t)MROWS * KPAD * 2);
    float*  xg   = (float*) alloc((size_t)MROWS * NX * 4);
    __bf16* hb0  = (__bf16*)alloc((size_t)MROWS * HID * 2);
    __bf16* hb1  = (__bf16*)alloc((size_t)MROWS * HID * 2);
    float*  cb0  = (float*) alloc((size_t)MROWS * HID * 4);
    float*  cb1  = (float*) alloc((size_t)MROWS * HID * 4);
    __bf16* hbuf[2] = { hb0, hb1 };
    float*  cbuf[2] = { cb0, cb1 };

    // weights -> bf16 (65536 + 2*81920 = 229376 elements)
    prep_weights_kernel<<<(229376 + 255) / 256, 256, 0, stream>>>(
        Wx_f, Ul_f, Ur_f, Wx, Ul, Ur);

    // embedding gather -> bf16 A (65536*128 elements)
    gather_embed_kernel<<<(MROWS * KPAD) / 256, 256, 0, stream>>>(emb, seq, Abf);

    // xg GEMM: grid (M/16, 2), 4 waves/block each computing 16x64
    xg_gemm_kernel<<<dim3(MROWS / 16, 2), 128, 0, stream>>>(Abf, Wx, bx, xg);

    // level 0 (elementwise)
    level0_kernel<<<(MROWS * HID) / 256, 256, 0, stream>>>(xg, bl, br, hbuf[0], cbuf[0]);

    // levels 1..12
    int cur = 0;
    for (int l = 1; l <= 12; ++l) {
        const int n      = SEQ >> l;
        const int n_prev = SEQ >> (l - 1);
        float* fo = (l == 12) ? out : nullptr;
        tree_level_kernel<<<n, 320, 0, stream>>>(
            hbuf[cur], cbuf[cur], hbuf[cur ^ 1], cbuf[cur ^ 1],
            xg, deps + (size_t)(l - 1) * (SEQ / 2),
            Ul, Ur, bl, br, n, n_prev, fo);
        cur ^= 1;
    }
}
